// EntityNLM_74156905333549
// MI455X (gfx1250) — compile-verified, matrix operand order checked by reading, not run on hardware
//
#include <hip/hip_runtime.h>
#include <hip/hip_bf16.h>
#include <math.h>

#define T_SEQ   2048
#define VOCAB   50257
#define DIMH    256
#define G4H     1024
#define NENT    64
#define ROW_OUT 50348      // 2 + 64 + 25 + 50257
#define PX_OFF  91
#define VPAD    50304      // 393 * 128 (pad vocab rows for 128-tall tiles)
#define EPAD    257        // LDS row stride for ents/P (bank-conflict-free)

typedef __attribute__((ext_vector_type(16))) __bf16 v16bf;
typedef __attribute__((ext_vector_type(8)))  __bf16 v8bf;
typedef __attribute__((ext_vector_type(8)))  float  v8f;

union Frag16 { v16bf v; v8bf h[2]; };

static __device__ __forceinline__ unsigned short f2bf(float f) {
  unsigned int u = __float_as_uint(f);
  u += 0x7fffu + ((u >> 16) & 1u);
  return (unsigned short)(u >> 16);
}
static __device__ __forceinline__ float sigm(float x) { return 1.0f / (1.0f + expf(-x)); }

// ---------------- Phase 0a: out_W (f32) -> bf16, zero-padded to VPAD rows ----
__global__ __launch_bounds__(256) void k_conv_w(const float* __restrict__ out_W,
                                                unsigned short* __restrict__ Wb) {
  int idx = blockIdx.x * 256 + threadIdx.x;   // grid = VPAD blocks
  int r = idx >> 8, k = idx & 255;
  float v = (r < VOCAB) ? out_W[(size_t)r * DIMH + k] : 0.0f;
  Wb[idx] = f2bf(v);
}

// ---------------- Phase 0b: G[t][r] = W_ih @ emb[tok_t] + b_ih (f32) ---------
__global__ __launch_bounds__(256) void k_gih(const int* __restrict__ tokens,
                                             const float* __restrict__ emb,
                                             const float* __restrict__ W_ih,
                                             const float* __restrict__ b_ih,
                                             float* __restrict__ G) {
  __shared__ __align__(16) float x[DIMH];
  int t = blockIdx.x;
  int tok = tokens[t];
  x[threadIdx.x] = emb[(size_t)tok * DIMH + threadIdx.x];
  __syncthreads();
  for (int r = threadIdx.x; r < G4H; r += 256) {
    const float4* w = (const float4*)(W_ih + (size_t)r * DIMH);
    const float4* xv4 = (const float4*)x;
    float acc = b_ih[r];
#pragma unroll 8
    for (int k = 0; k < DIMH / 4; ++k) {
      float4 wv = w[k]; float4 xv = xv4[k];
      acc += wv.x * xv.x + wv.y * xv.y + wv.z * xv.z + wv.w * xv.w;
    }
    G[(size_t)t * G4H + r] = acc;
  }
}

// ---------------- Phase 1: sequential recurrence (single workgroup) ----------
__global__ __launch_bounds__(1024) void k_seq(
    const int* __restrict__ entity_ids,
    const float* __restrict__ G,
    const float* __restrict__ W_hh,  const float* __restrict__ b_hh,
    const float* __restrict__ r_emb,
    const float* __restrict__ Wr_W,  const float* __restrict__ Wr_b,
    const float* __restrict__ Wlen_W,const float* __restrict__ Wlen_b,
    const float* __restrict__ Went_W,const float* __restrict__ Went_b,
    const float* __restrict__ wdist_W,const float* __restrict__ wdist_b,
    const float* __restrict__ Wdelta_W,const float* __restrict__ Wdelta_b,
    const float* __restrict__ We_W,  const float* __restrict__ We_b,
    const float* __restrict__ ents0, const float* __restrict__ dist0,
    float* __restrict__ out, unsigned short* __restrict__ Vb) {
  extern __shared__ float smem[];
  float* sh_h    = smem;               // 256
  float* sh_c    = sh_h + 256;         // 256
  float* sh_g    = sh_c + 256;         // 1024
  float* sh_q1   = sh_g + 1024;        // 256
  float* sh_q2   = sh_q1 + 256;        // 256
  float* sh_u    = sh_q2 + 256;        // 256
  float* sh_dist = sh_u + 256;         // 64
  float* sh_rs   = sh_dist + 64;       // 64
  float* sh_sc   = sh_rs + 64;         // 8
  float* sh_ents = sh_sc + 8;          // 64*257
  float* sh_P    = sh_ents + NENT*EPAD;// 64*257

  const int tid = threadIdx.x;
  // ---- init carry state ----
  if (tid < 256) { sh_h[tid] = 0.0f; sh_c[tid] = 0.0f; }
  for (int p = tid; p < NENT * DIMH; p += 1024) {
    int e = p >> 8, k = p & 255;
    sh_ents[e * EPAD + k] = ents0[p];
  }
  if (tid < NENT) sh_dist[tid] = dist0[tid];
  __syncthreads();
  // P = ents @ Went  (one-time full build; afterwards incremental)
  for (int p = tid; p < NENT * DIMH; p += 1024) {
    int e = p >> 8, j = p & 255;
    float acc = 0.0f;
    for (int k = 0; k < DIMH; ++k) acc += sh_ents[e * EPAD + k] * Went_W[(size_t)k * DIMH + j];
    sh_P[e * EPAD + j] = acc;
  }
  __syncthreads();
  if (tid < NENT) {
    float s = 0.0f;
    for (int j = 0; j < DIMH; ++j) s += sh_P[tid * EPAD + j];
    sh_rs[tid] = s;
  }
  __syncthreads();

  const float wdw = wdist_W[0], wdb = wdist_b[0], gwb = Went_b[0];
  for (int t = 0; t < T_SEQ; ++t) {
    const int eid = entity_ids[t];
    const float* ecur = sh_ents + eid * EPAD;
    float* orow = out + (size_t)t * ROW_OUT;
    // gates = G[t] + b_hh + W_hh @ h   (1024 rows, one per thread)
    {
      float acc = G[(size_t)t * G4H + tid] + b_hh[tid];
      const float4* w = (const float4*)(W_hh + (size_t)tid * DIMH);
#pragma unroll 8
      for (int k = 0; k < DIMH / 4; ++k) {
        float4 wv = w[k];
        acc += wv.x * sh_h[4*k] + wv.y * sh_h[4*k+1] + wv.z * sh_h[4*k+2] + wv.w * sh_h[4*k+3];
      }
      sh_g[tid] = acc;
    }
    __syncthreads();
    if (tid < 256) {   // LSTM cell (torch gate order i,f,g,o)
      float ii = sigm(sh_g[tid]);
      float ff = sigm(sh_g[256 + tid]);
      float gg = tanhf(sh_g[512 + tid]);
      float oo = sigm(sh_g[768 + tid]);
      float c = ff * sh_c[tid] + ii * gg;
      sh_c[tid] = c;
      sh_h[tid] = oo * tanhf(c);
    }
    __syncthreads();
    // parallel matvec phase
    if (tid < 256) {                       // q1 = Wr_W @ h
      const float4* w = (const float4*)(Wr_W + (size_t)tid * DIMH);
      float acc = 0.0f;
#pragma unroll 8
      for (int k = 0; k < DIMH / 4; ++k) {
        float4 wv = w[k];
        acc += wv.x * sh_h[4*k] + wv.y * sh_h[4*k+1] + wv.z * sh_h[4*k+2] + wv.w * sh_h[4*k+3];
      }
      sh_q1[tid] = acc;
    } else if (tid < 512) {                // q2 = Wdelta_W @ h
      int j = tid - 256;
      const float4* w = (const float4*)(Wdelta_W + (size_t)j * DIMH);
      float acc = 0.0f;
#pragma unroll 8
      for (int k = 0; k < DIMH / 4; ++k) {
        float4 wv = w[k];
        acc += wv.x * sh_h[4*k] + wv.y * sh_h[4*k+1] + wv.z * sh_h[4*k+2] + wv.w * sh_h[4*k+3];
      }
      sh_q2[j] = acc;
    } else if (tid < 768) {                // v = h + We_W@e_cur + We_b -> bf16
      int j = tid - 512;
      const float4* w = (const float4*)(We_W + (size_t)j * DIMH);
      float acc = We_b[j];
#pragma unroll 8
      for (int k = 0; k < DIMH / 4; ++k) {
        float4 wv = w[k];
        acc += wv.x * ecur[4*k] + wv.y * ecur[4*k+1] + wv.z * ecur[4*k+2] + wv.w * ecur[4*k+3];
      }
      Vb[(size_t)t * DIMH + j] = f2bf(sh_h[j] + acc);
    } else if (tid < 768 + NENT) {         // pred_e via incremental P
      int e = tid - 768;
      float acc = 0.0f;
      for (int k = 0; k < DIMH; ++k) acc += sh_P[e * EPAD + k] * sh_h[k];
      float d = sh_dist[e] - (float)t;
      orow[2 + e] = acc + (d * wdw + wdb) * sh_rs[e] + gwb;
    } else if (tid < 768 + NENT + 25) {    // pred_l
      int l = tid - 768 - NENT;
      const float* w = Wlen_W + (size_t)l * 512;
      float acc = Wlen_b[l];
      for (int k = 0; k < DIMH; ++k) acc += w[k] * sh_h[k];
      for (int k = 0; k < DIMH; ++k) acc += w[256 + k] * ecur[k];
      orow[66 + l] = acc;
    }
    __syncthreads();
    if (tid == 0) {
      float acc = Wr_b[0];
      for (int k = 0; k < DIMH; ++k) acc += r_emb[k] * sh_q1[k];
      orow[0] = acc;
    } else if (tid == 1) {
      float acc = Wr_b[0];
      for (int k = 0; k < DIMH; ++k) acc += r_emb[256 + k] * sh_q1[k];
      orow[1] = acc;
    } else if (tid == 2) {
      float acc = Wdelta_b[0];
      for (int k = 0; k < DIMH; ++k) acc += ecur[k] * sh_q2[k];
      sh_sc[0] = sigm(acc);
    }
    __syncthreads();
    if (tid < 256) {   // u = delta*e + (1-delta)*h
      float delta = sh_sc[0];
      sh_u[tid] = delta * ecur[tid] + (1.0f - delta) * sh_h[tid];
    }
    __syncthreads();
    if (tid < 32) {    // ||u||
      float p = 0.0f;
#pragma unroll
      for (int i = 0; i < 8; ++i) { float xv = sh_u[tid * 8 + i]; p += xv * xv; }
      for (int off = 16; off; off >>= 1) p += __shfl_down(p, off);
      if (tid == 0) sh_sc[1] = 1.0f / sqrtf(p);
    }
    __syncthreads();
    if (tid < 256) {   // scatter normalized entity row
      float un = sh_u[tid] * sh_sc[1];
      sh_u[tid] = un;
      sh_ents[eid * EPAD + tid] = un;
    }
    __syncthreads();
    if (tid < 256) {   // incremental P row update: P[eid] = u_n @ Went
      float acc = 0.0f;
      for (int k = 0; k < DIMH; ++k) acc += sh_u[k] * Went_W[(size_t)k * DIMH + tid];
      sh_P[eid * EPAD + tid] = acc;
    }
    __syncthreads();
    if (tid < 32) {
      float p = 0.0f;
#pragma unroll
      for (int i = 0; i < 8; ++i) p += sh_P[eid * EPAD + tid * 8 + i];
      for (int off = 16; off; off >>= 1) p += __shfl_down(p, off);
      if (tid == 0) { sh_rs[eid] = p; sh_dist[eid] = (float)t; }
    }
    __syncthreads();
  }
}

// ---------------- Phase 2: WMMA bf16 GEMM: logits = out_W @ V ----------------
// Wave tile 64(M) x 32(N), 8 waves -> 128x128 block, K=256 unrolled.
__global__ __launch_bounds__(256) void k_gemm(const __bf16* __restrict__ Wb,
                                              const __bf16* __restrict__ Vb,
                                              const float* __restrict__ out_b,
                                              float* __restrict__ out) {
  const int lane = threadIdx.x & 31;
  const int w    = threadIdx.x >> 5;
  const int wm   = w & 1, wn = w >> 1;
  const int m0   = blockIdx.y * 128 + wm * 64;
  const int n0   = blockIdx.x * 128 + wn * 32;
  const int lrow = lane & 15;
  const int khalf = (lane >> 4) << 3;   // 0 or 8: K-half split across lane groups

  const __bf16* arow[4];
  const __bf16* brow[2];
#pragma unroll
  for (int i = 0; i < 4; ++i) arow[i] = Wb + (size_t)(m0 + i * 16 + lrow) * DIMH;
#pragma unroll
  for (int j = 0; j < 2; ++j) brow[j] = Vb + (size_t)(n0 + j * 16 + lrow) * DIMH;

  v8f acc[4][2];
#pragma unroll
  for (int i = 0; i < 4; ++i)
#pragma unroll
    for (int j = 0; j < 2; ++j) acc[i][j] = (v8f){0.f,0.f,0.f,0.f,0.f,0.f,0.f,0.f};

#pragma unroll
  for (int kk = 0; kk < DIMH; kk += 32) {
    const int kb = kk + khalf;
    Frag16 a[4], b[2];
#pragma unroll
    for (int i = 0; i < 4; ++i) {
      a[i].h[0] = *(const v8bf*)(arow[i] + kb);
      a[i].h[1] = *(const v8bf*)(arow[i] + kb + 16);
    }
#pragma unroll
    for (int j = 0; j < 2; ++j) {
      b[j].h[0] = *(const v8bf*)(brow[j] + kb);
      b[j].h[1] = *(const v8bf*)(brow[j] + kb + 16);
    }
#pragma unroll
    for (int i = 0; i < 4; ++i)
#pragma unroll
      for (int j = 0; j < 2; ++j)
        acc[i][j] = __builtin_amdgcn_wmma_f32_16x16x32_bf16(
            false, a[i].v, false, b[j].v, (short)0, acc[i][j], false, false);
  }

  const int rofs = (lane >> 4) << 3;  // C layout: lanes 16-31 hold M=r+8
#pragma unroll
  for (int i = 0; i < 4; ++i) {
#pragma unroll
    for (int j = 0; j < 2; ++j) {
      const int col = n0 + j * 16 + (lane & 15);
      const size_t cbase = (size_t)col * ROW_OUT + PX_OFF;
#pragma unroll
      for (int r = 0; r < 8; ++r) {
        const int row = m0 + i * 16 + rofs + r;
        if (row < VOCAB) out[cbase + row] = acc[i][j][r] + out_b[row];
      }
    }
  }
}

extern "C" void kernel_launch(void* const* d_in, const int* in_sizes, int n_in,
                              void* d_out, int out_size, void* d_ws, size_t ws_size,
                              hipStream_t stream) {
  const int*   tokens   = (const int*)d_in[0];
  const int*   eids     = (const int*)d_in[1];
  const float* emb      = (const float*)d_in[2];
  const float* W_ih     = (const float*)d_in[3];
  const float* W_hh     = (const float*)d_in[4];
  const float* b_ih     = (const float*)d_in[5];
  const float* b_hh     = (const float*)d_in[6];
  const float* out_W    = (const float*)d_in[7];
  const float* out_b    = (const float*)d_in[8];
  const float* r_emb    = (const float*)d_in[9];
  const float* Wr_W     = (const float*)d_in[10];
  const float* Wr_b     = (const float*)d_in[11];
  const float* Wlen_W   = (const float*)d_in[12];
  const float* Wlen_b   = (const float*)d_in[13];
  const float* Went_W   = (const float*)d_in[14];
  const float* Went_b   = (const float*)d_in[15];
  const float* wdist_W  = (const float*)d_in[16];
  const float* wdist_b  = (const float*)d_in[17];
  const float* Wdelta_W = (const float*)d_in[18];
  const float* Wdelta_b = (const float*)d_in[19];
  const float* We_W     = (const float*)d_in[20];
  const float* We_b     = (const float*)d_in[21];
  const float* ents0    = (const float*)d_in[22];
  const float* dist0    = (const float*)d_in[23];
  float* out = (float*)d_out;

  char* ws = (char*)d_ws;
  unsigned short* Wb = (unsigned short*)ws;                               // VPAD*256 bf16
  unsigned short* Vb = (unsigned short*)(ws + (size_t)VPAD * DIMH * 2);   // T*256 bf16
  float*          G  = (float*)(ws + (size_t)VPAD * DIMH * 2
                                    + (size_t)T_SEQ * DIMH * 2);          // T*1024 f32

  k_conv_w<<<VPAD, 256, 0, stream>>>(out_W, Wb);
  k_gih<<<T_SEQ, 256, 0, stream>>>(tokens, emb, W_ih, b_ih, G);

  const size_t smem_bytes =
      (size_t)(256 + 256 + 1024 + 256 + 256 + 256 + 64 + 64 + 8 + 2 * NENT * EPAD) * 4;
  k_seq<<<1, 1024, smem_bytes, stream>>>(eids, G, W_hh, b_hh, r_emb,
      Wr_W, Wr_b, Wlen_W, Wlen_b, Went_W, Went_b, wdist_W, wdist_b,
      Wdelta_W, Wdelta_b, We_W, We_b, ents0, dist0, out, Vb);

  k_gemm<<<dim3(T_SEQ / 128, VPAD / 128), 256, 0, stream>>>(
      (const __bf16*)Wb, (const __bf16*)Vb, out_b, out);
}
